// MixtureOfExpertsRotaryStftDecoder_21869973471565
// MI455X (gfx1250) — matrix operand order repaired and checked by reading, not compiled, not run on hardware
//
#include <hip/hip_runtime.h>

// ---------------------------------------------------------------------------
// MixtureOfExpertsRotaryStftDecoder for MI455X (gfx1250, wave32, WMMA).
// All matmuls run through v_wmma_f32_16x16x32_bf16, f32 accumulation.
// GEMM: 256 thr = 8 waves (2M x 4N), wave tile 32x32 (4 WMMAs/K-step),
// guard-free inner loop, 128-bit vector fragment loads.
// MoE uses top-2 sparse routing: per-expert token lists + gather/scatter GEMM
// (static worst-case grids with device-count early exit -> graph-capture safe).
// ---------------------------------------------------------------------------

#define B_     2
#define CIN_   512
#define S_     1024
#define D_     768
#define H_     12
#define HD_    64
#define E_     8
#define L_     2
#define FF_    2048
#define FBINS_ 513
#define N2_    1026          // 2*FBINS
#define KPAD_  1056          // N2_ padded to multiple of 32 for WMMA
#define WIN_   1024
#define HOP_   256
#define PADT_  384
#define OUTL_  262144        // (S-1)*HOP + WIN - 2*PAD
#define BS_    (B_*S_)

typedef __bf16 bf16_t;
typedef __attribute__((ext_vector_type(8)))  __bf16 bf16x8;
typedef __attribute__((ext_vector_type(16))) __bf16 bf16x16;
typedef __attribute__((ext_vector_type(8)))  float  f32x8;

struct GemmP {
    const bf16_t* A; const bf16_t* Bm; float* C; bf16_t* Cbf;
    const float* bias; const float* rowscale;
    const int* gatherA;   // if set: A row m -> gatherA[m] (clamped to 0 if <0)
    const int* scatterC;  // if set: C/Cbf/rowscale row m -> scatterC[m]; <0 skips store
    const int* cntPtr;    // if set: blocks with m0 >= *cntPtr exit (count mult of 64)
    int M, N, K, lda, ldb, ldc, ldcbf;
    long long sA1, sA2, sB1, sB2, sC1, sC2, sF1, sF2;
    int bdim2, rsStride, gelu, accum;
    float alpha;
};

__device__ __forceinline__ float gelu_tanh(float x) {
    return 0.5f * x * (1.0f + tanhf(0.7978845608028654f * (x + 0.044715f * x * x * x)));
}

__device__ __forceinline__ bf16x16 cat16(bf16x8 lo, bf16x8 hi) {
    return __builtin_shufflevector(lo, hi, 0, 1, 2, 3, 4, 5, 6, 7,
                                           8, 9, 10, 11, 12, 13, 14, 15);
}

// A fragment (16x32 bf16): lane = row m (lo), half-wave hi selects K slices
//   a[0..7] = A[m][kk + hi*8 .. +7], a[8..15] = A[m][kk + 16 + hi*8 .. +7]
// B fragment (32x16 bf16): lane = col N (lo), b[0..7] = B[kk + hi*8 .. +7][N], b[8..15] = B[kk + 16 + hi*8 .. +7][N]
// C/D: VGPR i -> M = i + 8*hi, N = n0 + (lane&15)
__global__ __launch_bounds__(256) void wmma_gemm(GemmP p) {
    int z  = blockIdx.z;
    int z1 = z / p.bdim2, z2 = z - z1 * p.bdim2;
    const bf16_t* Ab = p.A  + z1 * p.sA1 + z2 * p.sA2;
    const bf16_t* Bb = p.Bm + z1 * p.sB1 + z2 * p.sB2;

    int wave = threadIdx.x >> 5, lane = threadIdx.x & 31;
    int wm = wave & 1, wn = wave >> 1;
    int m0 = blockIdx.y * 64  + wm * 32;    // M is always a multiple of 64
    int n0 = blockIdx.x * 128 + wn * 32;
    if (n0 >= p.N) return;                   // wave-uniform (N multiple of 32)
    if (p.cntPtr && m0 >= *p.cntPtr) return; // sparse-MoE early exit (uniform)

    int hi = lane >> 4, lo = lane & 15;

    long long arow0, arow1;
    if (p.gatherA) {
        int r0 = p.gatherA[m0 + lo];      if (r0 < 0) r0 = 0;
        int r1 = p.gatherA[m0 + 16 + lo]; if (r1 < 0) r1 = 0;
        arow0 = r0; arow1 = r1;
    } else {
        arow0 = m0 + lo; arow1 = m0 + 16 + lo;
    }

    f32x8 acc00, acc01, acc10, acc11;
    #pragma unroll
    for (int i = 0; i < 8; i++) { acc00[i] = 0.f; acc01[i] = 0.f; acc10[i] = 0.f; acc11[i] = 0.f; }

    const bf16_t* Ar0 = Ab + arow0 * p.lda + hi * 8;
    const bf16_t* Ar1 = Ab + arow1 * p.lda + hi * 8;
    const bf16_t* Br  = Bb + (long long)(hi * 8) * p.ldb + n0 + lo;   // col N = lo, rows K = 8*hi + {0..7, 16..23}

    for (int kk = 0; kk < p.K; kk += 32) {
        bf16x16 a0 = cat16(*(const bf16x8*)(Ar0),      *(const bf16x8*)(Ar0 + 16));
        bf16x16 a1 = cat16(*(const bf16x8*)(Ar1),      *(const bf16x8*)(Ar1 + 16));
        bf16x16 b0, b1;
        #pragma unroll
        for (int s = 0; s < 16; s++) {
            long long ko = (long long)((s < 8) ? s : s + 8) * p.ldb;
            b0[s] = Br[ko];
            b1[s] = Br[ko + 16];
        }
        acc00 = __builtin_amdgcn_wmma_f32_16x16x32_bf16(false, a0, false, b0, (short)0, acc00, false, false);
        acc01 = __builtin_amdgcn_wmma_f32_16x16x32_bf16(false, a0, false, b1, (short)0, acc01, false, false);
        acc10 = __builtin_amdgcn_wmma_f32_16x16x32_bf16(false, a1, false, b0, (short)0, acc10, false, false);
        acc11 = __builtin_amdgcn_wmma_f32_16x16x32_bf16(false, a1, false, b1, (short)0, acc11, false, false);
        Ar0 += 32;
        Ar1 += 32;
        Br  += (long long)32 * p.ldb;
    }

    long long cO = z1 * p.sC1 + z2 * p.sC2;
    long long fO = z1 * p.sF1 + z2 * p.sF2;
    #pragma unroll
    for (int mf = 0; mf < 2; mf++) {
        #pragma unroll
        for (int nf = 0; nf < 2; nf++) {
            const f32x8& c = mf ? (nf ? acc11 : acc10) : (nf ? acc01 : acc00);
            int ncol = n0 + nf * 16 + lo;
            #pragma unroll
            for (int i = 0; i < 8; i++) {
                int mi = m0 + mf * 16 + i + hi * 8;
                int crow = mi;
                bool ok = true;
                if (p.scatterC) { crow = p.scatterC[mi]; ok = (crow >= 0); }
                if (ok) {
                    float v = c[i] * p.alpha;
                    if (p.bias)     v += p.bias[ncol];
                    if (p.gelu)     v  = gelu_tanh(v);
                    if (p.rowscale) v *= p.rowscale[(long long)crow * p.rsStride];
                    float tot = v;
                    if (p.C) {
                        float* cp = p.C + cO + (long long)crow * p.ldc + ncol;
                        if (p.accum) tot += *cp;
                        *cp = tot;
                    }
                    if (p.Cbf) p.Cbf[fO + (long long)crow * p.ldcbf + ncol] = (bf16_t)tot;
                }
            }
        }
    }
}

// ---------------------------------------------------------------------------
// Glue kernels
// ---------------------------------------------------------------------------
__global__ void cast_f32_bf16(const float* in, bf16_t* out, long long n) {
    long long i = (long long)blockIdx.x * 256 + threadIdx.x;
    if (i < n) out[i] = (bf16_t)in[i];
}

// z[B,C,S] -> zT[B*S, C] (bf16)
__global__ void transpose_cast_z(const float* z, bf16_t* zT, long long total) {
    long long i = (long long)blockIdx.x * 256 + threadIdx.x;
    if (i >= total) return;
    int c = (int)(i % CIN_);
    long long bl = i / CIN_;
    int l = (int)(bl % S_);
    int b = (int)(bl / S_);
    zT[i] = (bf16_t)z[((long long)b * CIN_ + c) * S_ + l];
}

// k_bf[B*S, D] -> kT[b,h,d,s] so QK^T is a plain row-major GEMM
__global__ void transpose_k_heads(const bf16_t* k, bf16_t* kT, long long total) {
    long long i = (long long)blockIdx.x * 256 + threadIdx.x;
    if (i >= total) return;
    int s = (int)(i % S_);
    long long r = i / S_;
    int d = (int)(r % HD_);
    long long r2 = r / HD_;
    int h = (int)(r2 % H_);
    int b = (int)(r2 / H_);
    kT[i] = k[((long long)b * S_ + s) * D_ + h * HD_ + d];
}

// out_w [D, N2] -> zero-padded [D, KPAD] bf16 (pad cols -> 0 in s_bf directly)
__global__ void pad_cast_wout(const float* w, bf16_t* wp, long long total) {
    long long i = (long long)blockIdx.x * 256 + threadIdx.x;
    if (i >= total) return;
    int col = (int)(i % KPAD_);
    int row = (int)(i / KPAD_);
    wp[i] = (col < N2_) ? (bf16_t)w[(long long)row * N2_ + col] : (bf16_t)0.0f;
}

__global__ void pad_bias_f32(const float* b, float* bp) {
    int i = blockIdx.x * 256 + threadIdx.x;
    if (i < KPAD_) bp[i] = (i < N2_) ? b[i] : 0.0f;
}

__global__ __launch_bounds__(256) void layernorm_k(const float* x, const float* g,
                                                   const float* bta, bf16_t* out) {
    int row = blockIdx.x;
    const float* xr = x + (long long)row * D_;
    float s = 0.f, s2 = 0.f;
    for (int d = threadIdx.x; d < D_; d += 256) { float v = xr[d]; s += v; s2 += v * v; }
    __shared__ float r1[256], r2[256];
    r1[threadIdx.x] = s; r2[threadIdx.x] = s2;
    __syncthreads();
    for (int off = 128; off > 0; off >>= 1) {
        if (threadIdx.x < off) { r1[threadIdx.x] += r1[threadIdx.x + off];
                                 r2[threadIdx.x] += r2[threadIdx.x + off]; }
        __syncthreads();
    }
    float mean = r1[0] / D_;
    float var  = r2[0] / D_ - mean * mean;
    float rstd = rsqrtf(var + 1e-5f);
    for (int d = threadIdx.x; d < D_; d += 256)
        out[(long long)row * D_ + d] = (bf16_t)((xr[d] - mean) * rstd * g[d] + bta[d]);
}

// interleaved-pair RoPE over full hidden dim; reads f32, writes bf16
__global__ void rope_k(const float* q, bf16_t* qb, long long total) {
    long long i = (long long)blockIdx.x * 256 + threadIdx.x;
    if (i >= total) return;
    const int halfd = D_ / 2;
    int j = (int)(i % halfd);
    long long bs = i / halfd;
    int spos = (int)(bs % S_);
    float inv = powf(10000.0f, -(float)j / (float)halfd);
    float sn, cs;
    sincosf((float)spos * inv, &sn, &cs);
    long long base = bs * D_ + 2 * j;
    float x1 = q[base], x2 = q[base + 1];
    qb[base]     = (bf16_t)(x1 * cs - x2 * sn);
    qb[base + 1] = (bf16_t)(x1 * sn + x2 * cs);
}

// causal row softmax; f32 scores -> bf16 probabilities (feeds AV WMMA GEMM)
__global__ __launch_bounds__(256) void softmax_causal_k(const float* sc, bf16_t* att) {
    int row = blockIdx.x;               // 0 .. B*H*S-1
    int r = row % S_;
    const float* sr = sc + (long long)row * S_;
    bf16_t* ar = att + (long long)row * S_;
    __shared__ float red[256];
    float mx = -3.0e38f;
    for (int c0 = threadIdx.x; c0 < S_; c0 += 256)
        if (c0 <= r) mx = fmaxf(mx, sr[c0]);
    red[threadIdx.x] = mx; __syncthreads();
    for (int off = 128; off > 0; off >>= 1) {
        if (threadIdx.x < off) red[threadIdx.x] = fmaxf(red[threadIdx.x], red[threadIdx.x + off]);
        __syncthreads();
    }
    mx = red[0]; __syncthreads();
    float sum = 0.f;
    for (int c0 = threadIdx.x; c0 < S_; c0 += 256)
        if (c0 <= r) sum += expf(sr[c0] - mx);
    red[threadIdx.x] = sum; __syncthreads();
    for (int off = 128; off > 0; off >>= 1) {
        if (threadIdx.x < off) red[threadIdx.x] += red[threadIdx.x + off];
        __syncthreads();
    }
    float inv = 1.0f / red[0];
    for (int c0 = threadIdx.x; c0 < S_; c0 += 256)
        ar[c0] = (bf16_t)((c0 <= r) ? expf(sr[c0] - mx) * inv : 0.0f);
}

__global__ void zero_cnt_k(int* cnt) {
    if (threadIdx.x < E_) cnt[threadIdx.x] = 0;
}

// per-token gate logits + top-2 softmax; writes dense gate[BS,E] AND appends
// the token to its two experts' lists (order-independent final output)
__global__ __launch_bounds__(256) void moe_gate_k(const bf16_t* h, const float* gw,
                                                  float* gate, int* cnt, int* list) {
    int m = blockIdx.x;
    float acc[E_];
    #pragma unroll
    for (int e = 0; e < E_; e++) acc[e] = 0.f;
    for (int d = threadIdx.x; d < D_; d += 256) {
        float hv = (float)h[(long long)m * D_ + d];
        #pragma unroll
        for (int e = 0; e < E_; e++) acc[e] += hv * gw[d * E_ + e];
    }
    __shared__ float red[256 * E_];
    #pragma unroll
    for (int e = 0; e < E_; e++) red[threadIdx.x * E_ + e] = acc[e];
    __syncthreads();
    for (int off = 128; off > 0; off >>= 1) {
        if (threadIdx.x < off) {
            #pragma unroll
            for (int e = 0; e < E_; e++)
                red[threadIdx.x * E_ + e] += red[(threadIdx.x + off) * E_ + e];
        }
        __syncthreads();
    }
    if (threadIdx.x == 0) {
        float lg[E_];
        #pragma unroll
        for (int e = 0; e < E_; e++) lg[e] = red[e];
        int i0 = 0;
        for (int e = 1; e < E_; e++) if (lg[e] > lg[i0]) i0 = e;
        int i1 = -1;
        for (int e = 0; e < E_; e++) { if (e == i0) continue; if (i1 < 0 || lg[e] > lg[i1]) i1 = e; }
        float e1 = expf(lg[i1] - lg[i0]);
        float w0 = 1.0f / (1.0f + e1);
        #pragma unroll
        for (int e = 0; e < E_; e++) gate[(long long)m * E_ + e] = 0.0f;
        gate[(long long)m * E_ + i0] = w0;
        gate[(long long)m * E_ + i1] = e1 * w0;
        int p0 = atomicAdd(&cnt[i0], 1);
        list[i0 * BS_ + p0] = m;
        int p1 = atomicAdd(&cnt[i1], 1);
        list[i1 * BS_ + p1] = m;
    }
}

// pad each expert list to a multiple of 64 with -1; publish rounded counts
__global__ void pad_lists_k(const int* cnt, int* cntR, int* list) {
    int e = blockIdx.x;
    int c = cnt[e];
    int r = (c + 63) & ~63;
    if (threadIdx.x == 0) cntR[e] = r;
    for (int j = c + threadIdx.x; j < r; j += 64) list[e * BS_ + j] = -1;
}

// irfft basis [KPAD x WIN] (bf16): rows 0..512 = w_f*cos/N, 513..1025 = -2*sin/N
__global__ void istft_basis_k(bf16_t* basis) {
    long long i = (long long)blockIdx.x * 256 + threadIdx.x;
    const long long total = (long long)KPAD_ * WIN_;
    if (i >= total) return;
    int k = (int)(i / WIN_), n = (int)(i % WIN_);
    float v = 0.0f;
    if (k < FBINS_) {
        int f = k;
        float w = (f == 0 || f == FBINS_ - 1) ? 1.0f : 2.0f;
        int ph = (f * n) & (WIN_ - 1);
        v = w * cosf(6.283185307179586f * (float)ph / (float)WIN_) / (float)WIN_;
    } else if (k < 2 * FBINS_) {
        int f = k - FBINS_;
        if (f != 0 && f != FBINS_ - 1) {
            int ph = (f * n) & (WIN_ - 1);
            v = -2.0f * sinf(6.283185307179586f * (float)ph / (float)WIN_) / (float)WIN_;
        }
    }
    basis[i] = (bf16_t)v;
}

// overlap-add with periodic-Hann window and analytic envelope; trims PAD
__global__ void istft_ola_k(const float* frames, float* out) {
    long long i = (long long)blockIdx.x * 256 + threadIdx.x;
    const long long total = (long long)B_ * OUTL_;
    if (i >= total) return;
    int b  = (int)(i / OUTL_);
    int pp = (int)(i % OUTL_);
    int p  = pp + PADT_;
    int tmin = (p - (WIN_ - 1) > 0) ? ((p - (WIN_ - 1) + HOP_ - 1) / HOP_) : 0;
    int tmax = p / HOP_; if (tmax > S_ - 1) tmax = S_ - 1;
    float acc = 0.f, env = 0.f;
    for (int t = tmin; t <= tmax; t++) {
        int j = p - t * HOP_;
        float w = 0.5f * (1.0f - cosf(6.283185307179586f * (float)j / (float)WIN_));
        acc += frames[((long long)b * S_ + t) * WIN_ + j] * w;
        env += w * w;
    }
    out[i] = acc / env;
}

// ---------------------------------------------------------------------------
// Host orchestration
// ---------------------------------------------------------------------------
extern "C" void kernel_launch(void* const* d_in, const int* in_sizes, int n_in,
                              void* d_out, int out_size, void* d_ws, size_t ws_size,
                              hipStream_t stream) {
    (void)in_sizes; (void)n_in; (void)out_size; (void)ws_size;
    const float* z      = (const float*)d_in[0];
    const float* in_w   = (const float*)d_in[1];
    const float* in_b   = (const float*)d_in[2];
    const float* ln1_g  = (const float*)d_in[3];
    const float* ln1_b  = (const float*)d_in[4];
    const float* wq     = (const float*)d_in[5];
    const float* bq     = (const float*)d_in[6];
    const float* wk     = (const float*)d_in[7];
    const float* bk     = (const float*)d_in[8];
    const float* wv     = (const float*)d_in[9];
    const float* bv     = (const float*)d_in[10];
    const float* wo     = (const float*)d_in[11];
    const float* bo     = (const float*)d_in[12];
    const float* ln2_g  = (const float*)d_in[13];
    const float* ln2_b  = (const float*)d_in[14];
    const float* gate_w = (const float*)d_in[15];
    const float* e_w1   = (const float*)d_in[16];
    const float* e_b1   = (const float*)d_in[17];
    const float* e_w2   = (const float*)d_in[18];
    const float* e_b2   = (const float*)d_in[19];
    const float* out_w  = (const float*)d_in[20];
    const float* out_b  = (const float*)d_in[21];
    float* y_out = (float*)d_out;

    char* base = (char*)d_ws;
    size_t off = 0;
    auto alloc = [&](size_t bytes) -> void* {
        void* p = base + off;
        off = (off + bytes + 255) & ~(size_t)255;
        return p;
    };
    bf16_t* zT_bf    = (bf16_t*)alloc(sizeof(bf16_t) * (size_t)BS_ * CIN_);
    bf16_t* w_in_bf  = (bf16_t*)alloc(sizeof(bf16_t) * (size_t)CIN_ * D_);
    bf16_t* wq_bf    = (bf16_t*)alloc(sizeof(bf16_t) * (size_t)L_ * D_ * D_);
    bf16_t* wk_bf    = (bf16_t*)alloc(sizeof(bf16_t) * (size_t)L_ * D_ * D_);
    bf16_t* wv_bf    = (bf16_t*)alloc(sizeof(bf16_t) * (size_t)L_ * D_ * D_);
    bf16_t* wo_bf    = (bf16_t*)alloc(sizeof(bf16_t) * (size_t)L_ * D_ * D_);
    bf16_t* w1_bf    = (bf16_t*)alloc(sizeof(bf16_t) * (size_t)L_ * E_ * D_ * FF_);
    bf16_t* w2_bf    = (bf16_t*)alloc(sizeof(bf16_t) * (size_t)L_ * E_ * FF_ * D_);
    bf16_t* woutp_bf = (bf16_t*)alloc(sizeof(bf16_t) * (size_t)D_ * KPAD_);
    float*  outb_pad = (float*) alloc(sizeof(float)  * (size_t)KPAD_);
    bf16_t* basis_bf = (bf16_t*)alloc(sizeof(bf16_t) * (size_t)KPAD_ * WIN_);
    float*  x        = (float*) alloc(sizeof(float)  * (size_t)BS_ * D_);
    bf16_t* h_bf     = (bf16_t*)alloc(sizeof(bf16_t) * (size_t)BS_ * D_);
    float*  qf       = (float*) alloc(sizeof(float)  * (size_t)BS_ * D_);
    float*  kf       = (float*) alloc(sizeof(float)  * (size_t)BS_ * D_);
    bf16_t* q_bf     = (bf16_t*)alloc(sizeof(bf16_t) * (size_t)BS_ * D_);
    bf16_t* k_bf     = (bf16_t*)alloc(sizeof(bf16_t) * (size_t)BS_ * D_);
    bf16_t* kT_bf    = (bf16_t*)alloc(sizeof(bf16_t) * (size_t)BS_ * D_);
    bf16_t* v_bf     = (bf16_t*)alloc(sizeof(bf16_t) * (size_t)BS_ * D_);
    float*  scores   = (float*) alloc(sizeof(float)  * (size_t)B_ * H_ * S_ * S_);
    bf16_t* att_bf   = (bf16_t*)alloc(sizeof(bf16_t) * (size_t)B_ * H_ * S_ * S_);
    bf16_t* o_bf     = (bf16_t*)alloc(sizeof(bf16_t) * (size_t)BS_ * D_);
    float*  gate     = (float*) alloc(sizeof(float)  * (size_t)BS_ * E_);
    int*    ecnt     = (int*)   alloc(sizeof(int) * E_);
    int*    ecntR    = (int*)   alloc(sizeof(int) * E_);
    int*    elist    = (int*)   alloc(sizeof(int) * (size_t)E_ * BS_);
    bf16_t* h1_bf    = (bf16_t*)alloc(sizeof(bf16_t) * (size_t)BS_ * FF_);
    bf16_t* s_bf     = (bf16_t*)alloc(sizeof(bf16_t) * (size_t)BS_ * KPAD_);
    float*  frames   = (float*) alloc(sizeof(float)  * (size_t)BS_ * WIN_);

    auto cast = [&](const float* src, bf16_t* dst, long long n) {
        cast_f32_bf16<<<(unsigned)((n + 255) / 256), 256, 0, stream>>>(src, dst, n);
    };
    auto mk = [&]() -> GemmP {
        GemmP p{}; p.bdim2 = 1; p.alpha = 1.0f; p.rsStride = 1; return p;
    };
    auto gemm = [&](const GemmP& p, int batch) {
        dim3 g((unsigned)((p.N + 127) / 128), (unsigned)(p.M / 64), (unsigned)batch);
        wmma_gemm<<<g, 256, 0, stream>>>(p);
    };

    // ---- per-call prep (deterministic) ----
    cast(in_w, w_in_bf, (long long)CIN_ * D_);
    cast(wq,   wq_bf,   (long long)L_ * D_ * D_);
    cast(wk,   wk_bf,   (long long)L_ * D_ * D_);
    cast(wv,   wv_bf,   (long long)L_ * D_ * D_);
    cast(wo,   wo_bf,   (long long)L_ * D_ * D_);
    cast(e_w1, w1_bf,   (long long)L_ * E_ * D_ * FF_);
    cast(e_w2, w2_bf,   (long long)L_ * E_ * FF_ * D_);
    pad_cast_wout<<<(unsigned)(((long long)D_ * KPAD_ + 255) / 256), 256, 0, stream>>>(
        out_w, woutp_bf, (long long)D_ * KPAD_);
    pad_bias_f32<<<(KPAD_ + 255) / 256, 256, 0, stream>>>(out_b, outb_pad);
    istft_basis_k<<<(unsigned)(((long long)KPAD_ * WIN_ + 255) / 256), 256, 0, stream>>>(basis_bf);
    transpose_cast_z<<<(unsigned)(((long long)BS_ * CIN_ + 255) / 256), 256, 0, stream>>>(
        z, zT_bf, (long long)BS_ * CIN_);

    // ---- in-projection: x = zT @ in_w + in_b  [BS,D] ----
    { GemmP p = mk(); p.A = zT_bf; p.lda = CIN_; p.Bm = w_in_bf; p.ldb = D_;
      p.C = x; p.ldc = D_; p.bias = in_b; p.M = BS_; p.N = D_; p.K = CIN_; gemm(p, 1); }

    for (int i = 0; i < L_; i++) {
        bool use_rope = (i == 0);
        layernorm_k<<<BS_, 256, 0, stream>>>(x, ln1_g + i * D_, ln1_b + i * D_, h_bf);

        // q, k, v projections
        { GemmP p = mk(); p.A = h_bf; p.lda = D_;
          p.Bm = wq_bf + (long long)i * D_ * D_; p.ldb = D_; p.bias = bq + i * D_;
          p.M = BS_; p.N = D_; p.K = D_;
          if (use_rope) { p.C = qf; p.ldc = D_; } else { p.Cbf = q_bf; p.ldcbf = D_; }
          gemm(p, 1); }
        { GemmP p = mk(); p.A = h_bf; p.lda = D_;
          p.Bm = wk_bf + (long long)i * D_ * D_; p.ldb = D_; p.bias = bk + i * D_;
          p.M = BS_; p.N = D_; p.K = D_;
          if (use_rope) { p.C = kf; p.ldc = D_; } else { p.Cbf = k_bf; p.ldcbf = D_; }
          gemm(p, 1); }
        { GemmP p = mk(); p.A = h_bf; p.lda = D_;
          p.Bm = wv_bf + (long long)i * D_ * D_; p.ldb = D_; p.bias = bv + i * D_;
          p.Cbf = v_bf; p.ldcbf = D_; p.M = BS_; p.N = D_; p.K = D_; gemm(p, 1); }

        if (use_rope) {
            long long tot = (long long)BS_ * (D_ / 2);
            rope_k<<<(unsigned)((tot + 255) / 256), 256, 0, stream>>>(qf, q_bf, tot);
            rope_k<<<(unsigned)((tot + 255) / 256), 256, 0, stream>>>(kf, k_bf, tot);
        }
        transpose_k_heads<<<(unsigned)(((long long)BS_ * D_ + 255) / 256), 256, 0, stream>>>(
            k_bf, kT_bf, (long long)BS_ * D_);

        // scores[b,h] = 0.125 * q_h @ kT_h   (batched over B*H, plain GEMM)
        { GemmP p = mk(); p.A = q_bf; p.lda = D_; p.sA1 = (long long)S_ * D_; p.sA2 = HD_;
          p.Bm = kT_bf; p.ldb = S_;
          p.sB1 = (long long)H_ * HD_ * S_; p.sB2 = (long long)HD_ * S_;
          p.C = scores; p.ldc = S_; p.sC1 = (long long)H_ * S_ * S_; p.sC2 = (long long)S_ * S_;
          p.bdim2 = H_; p.alpha = 0.125f; p.M = S_; p.N = S_; p.K = HD_;
          gemm(p, B_ * H_); }

        softmax_causal_k<<<B_ * H_ * S_, 256, 0, stream>>>(scores, att_bf);

        // o[b,h] = att @ v_h, written head-interleaved into o_bf[BS,D]
        { GemmP p = mk(); p.A = att_bf; p.lda = S_;
          p.sA1 = (long long)H_ * S_ * S_; p.sA2 = (long long)S_ * S_;
          p.Bm = v_bf; p.ldb = D_; p.sB1 = (long long)S_ * D_; p.sB2 = HD_;
          p.Cbf = o_bf; p.ldcbf = D_; p.sF1 = (long long)S_ * D_; p.sF2 = HD_;
          p.bdim2 = H_; p.M = S_; p.N = HD_; p.K = S_;
          gemm(p, B_ * H_); }

        // x += o @ wo + bo   (residual accumulate in GEMM epilogue)
        { GemmP p = mk(); p.A = o_bf; p.lda = D_;
          p.Bm = wo_bf + (long long)i * D_ * D_; p.ldb = D_;
          p.C = x; p.ldc = D_; p.accum = 1; p.bias = bo + i * D_;
          p.M = BS_; p.N = D_; p.K = D_; gemm(p, 1); }

        // MoE: ln2 -> top-2 routing -> sparse gather/scatter expert GEMMs
        layernorm_k<<<BS_, 256, 0, stream>>>(x, ln2_g + i * D_, ln2_b + i * D_, h_bf);
        zero_cnt_k<<<1, 32, 0, stream>>>(ecnt);
        moe_gate_k<<<BS_, 256, 0, stream>>>(h_bf, gate_w + (long long)i * D_ * E_,
                                            gate, ecnt, elist);
        pad_lists_k<<<E_, 64, 0, stream>>>(ecnt, ecntR, elist);
        for (int e = 0; e < E_; e++) {
            // h1[slot] = gelu(h[token] @ w1[e] + b1[e])  (gathered rows)
            { GemmP p = mk(); p.A = h_bf; p.lda = D_;
              p.gatherA = elist + e * BS_; p.cntPtr = ecntR + e;
              p.Bm = w1_bf + ((long long)i * E_ + e) * D_ * FF_; p.ldb = FF_;
              p.Cbf = h1_bf; p.ldcbf = FF_;
              p.bias = e_b1 + ((long long)i * E_ + e) * FF_; p.gelu = 1;
              p.M = BS_; p.N = FF_; p.K = D_; gemm(p, 1); }
            // x[token] += gate[token,e] * (h1[slot] @ w2[e] + b2[e]) (scatter)
            { GemmP p = mk(); p.A = h1_bf; p.lda = FF_;
              p.scatterC = elist + e * BS_; p.cntPtr = ecntR + e;
              p.Bm = w2_bf + ((long long)i * E_ + e) * FF_ * D_; p.ldb = D_;
              p.C = x; p.ldc = D_; p.accum = 1;
              p.bias = e_b2 + ((long long)i * E_ + e) * D_;
              p.rowscale = gate + e; p.rsStride = E_;
              p.M = BS_; p.N = D_; p.K = FF_; gemm(p, 1); }
        }
    }

    // ---- out projection: s = x @ out_w_pad + out_b_pad -> bf16 K-padded buf
    cast(x, h_bf, (long long)BS_ * D_);  // reuse h_bf as x_bf
    { GemmP p = mk(); p.A = h_bf; p.lda = D_; p.Bm = woutp_bf; p.ldb = KPAD_;
      p.Cbf = s_bf; p.ldcbf = KPAD_; p.bias = outb_pad;
      p.M = BS_; p.N = KPAD_; p.K = D_; gemm(p, 1); }

    // ---- irfft as GEMM: frames[BS, WIN] = s_bf @ basis ----
    { GemmP p = mk(); p.A = s_bf; p.lda = KPAD_; p.Bm = basis_bf; p.ldb = WIN_;
      p.C = frames; p.ldc = WIN_; p.M = BS_; p.N = WIN_; p.K = KPAD_; gemm(p, 1); }

    // ---- windowed overlap-add + envelope normalize + trim ----
    istft_ola_k<<<(unsigned)(((long long)B_ * OUTL_ + 255) / 256), 256, 0, stream>>>(frames, y_out);
}
